// DecoderRNN_s_64338610094532
// MI455X (gfx1250) — matrix operand order repaired and checked by reading, not compiled, not run on hardware
//
#include <hip/hip_runtime.h>
#include <hip/hip_bf16.h>

// ---------------------------------------------------------------------------
// GRU decoder for gfx1250 (MI455X): bf16 WMMA GEMMs, async global->LDS
// staging (ASYNCcnt), persistent scan kernel with software grid barrier.
// ---------------------------------------------------------------------------

#define Vv   32000
#define Hd   1024
#define Tt   256
#define Bb   64
#define Gd   (3 * Hd)          // 3072
#define Mtot (Tt * Bb)         // 16384 rows of the batched input GEMM

// Flip to 0 if the async mnemonics are rejected by this toolchain's assembler.
#ifndef USE_ASYNC_LDS
#define USE_ASYNC_LDS 1
#endif

typedef __attribute__((ext_vector_type(16))) __bf16 v16bf;
typedef __attribute__((ext_vector_type(8)))  __bf16 v8bf;
typedef __attribute__((ext_vector_type(8)))  float  v8f;

// ------------------------- async global -> LDS staging ---------------------
// chunks16 x 16B, cooperatively copied by `nthreads` threads. Uses
// GLOBAL_LOAD_ASYNC_TO_LDS_B128 (ASYNCcnt) so no VGPR round-trip; caller must
// __syncthreads() afterwards before reading LDS.
__device__ inline void stage_to_lds(__bf16* lds, const __bf16* gsrc,
                                    int chunks16, int tid, int nthreads) {
#if USE_ASYNC_LDS
  const unsigned lbase = (unsigned)(size_t)lds;     // AS3 offset = low 32 bits
  for (int i = tid; i < chunks16; i += nthreads) {
    const unsigned  laddr = lbase + i * 16;
    const unsigned long long gaddr = (unsigned long long)(gsrc + i * 8);
    asm volatile("global_load_async_to_lds_b128 %0, %1, off"
                 :: "v"(laddr), "v"(gaddr) : "memory");
  }
  asm volatile("s_wait_asynccnt 0" ::: "memory");
#else
  const uint4* s = (const uint4*)gsrc;
  uint4*       d = (uint4*)lds;
  for (int i = tid; i < chunks16; i += nthreads) d[i] = s[i];
#endif
}

// ------------------------------ phase 1 ------------------------------------

__global__ __launch_bounds__(256) void k_embed_relu_bf16(
    const int* __restrict__ tokens, const float* __restrict__ emb,
    __bf16* __restrict__ x_bf) {
  const int m   = blockIdx.x;            // m = t*B + b
  const int tok = tokens[m];
  const float4 v = ((const float4*)(emb + (size_t)tok * Hd))[threadIdx.x];
  __bf16* dst = x_bf + (size_t)m * Hd + threadIdx.x * 4;
  dst[0] = (__bf16)fmaxf(v.x, 0.0f);
  dst[1] = (__bf16)fmaxf(v.y, 0.0f);
  dst[2] = (__bf16)fmaxf(v.z, 0.0f);
  dst[3] = (__bf16)fmaxf(v.w, 0.0f);
}

__global__ __launch_bounds__(256) void k_cvt_bf16(
    const float* __restrict__ src, __bf16* __restrict__ dst, int n4) {
  const int i = blockIdx.x * 256 + threadIdx.x;
  if (i < n4) {
    const float4 v = ((const float4*)src)[i];
    __bf16* d = dst + (size_t)i * 4;
    d[0] = (__bf16)v.x; d[1] = (__bf16)v.y; d[2] = (__bf16)v.z; d[3] = (__bf16)v.w;
  }
}

__global__ __launch_bounds__(256) void k_init_h(
    const float* __restrict__ h0, float* __restrict__ h,
    __bf16* __restrict__ h_bf, unsigned* __restrict__ bar) {
  const int i = blockIdx.x * 256 + threadIdx.x;
  const float v = h0[i];
  h[i]    = v;
  h_bf[i] = (__bf16)v;
  if (i < 4) bar[i] = 0u;
}

// ------------------------- WMMA fragment helpers ---------------------------
// A fragment (16x32 bf16 MxK): lane L = row (L&15); hi=L>>4 selects K-halves.
__device__ inline v16bf load_a_frag(const __bf16* __restrict__ row_base) {
  // row_base points at &A[m][k0 + hi*16]; elements {0..15}=K+0..15
  v8bf lo = *(const v8bf*)(row_base);
  v8bf hi = *(const v8bf*)(row_base + 8);
  return __builtin_shufflevector(lo, hi, 0, 1, 2, 3, 4, 5, 6, 7,
                                         8, 9, 10, 11, 12, 13, 14, 15);
}

// ------------------------------ phase 2 ------------------------------------
// xp[m,n] = sum_k x[m,k]*Wih[n,k] + b_ih[n]   (M=16384, N=3072, K=1024).
// Block: 256 threads (8 waves); 32-row A panel (64 KB) staged in LDS async.
// Each wave: 32x16 tile (2 M-tiles, B fragment reused, 2 indep. WMMA chains).
__global__ __launch_bounds__(256) void k_xp_gemm(
    const __bf16* __restrict__ x_bf, const __bf16* __restrict__ Wih,
    const float* __restrict__ b_ih, float* __restrict__ xp) {
  __shared__ __bf16 ldsA[32 * Hd];                    // 64 KB of 320 KB WGP LDS

  const int tid = threadIdx.x;
  const int m0  = blockIdx.x * 32;
  const int n0  = blockIdx.y * 128;

  stage_to_lds(ldsA, x_bf + (size_t)m0 * Hd, 32 * Hd / 8, tid, 256);
  __syncthreads();

  const int wave = tid >> 5;
  const int lane = tid & 31;
  const int lo16 = lane & 15;
  const int hi   = lane >> 4;

  const int n = n0 + wave * 16 + lo16;
  const __bf16* arow0 = ldsA + lo16 * Hd + hi * 16;          // M-tile 0
  const __bf16* arow1 = ldsA + (16 + lo16) * Hd + hi * 16;   // M-tile 1
  const __bf16* brow  = Wih + (size_t)n * Hd + hi * 16;

  v8f acc0 = {}, acc1 = {};
#pragma unroll 4
  for (int k0 = 0; k0 < Hd; k0 += 32) {
    __builtin_prefetch(brow + k0 + 512, 0, 1);        // global_prefetch_b8
    v16bf b  = *(const v16bf*)(brow + k0);            // shared by both tiles
    v16bf a0 = load_a_frag(arow0 + k0);
    v16bf a1 = load_a_frag(arow1 + k0);
    acc0 = __builtin_amdgcn_wmma_f32_16x16x32_bf16(false, a0, false, b,
                                                   (short)0, acc0, false, false);
    acc1 = __builtin_amdgcn_wmma_f32_16x16x32_bf16(false, a1, false, b,
                                                   (short)0, acc1, false, false);
  }

  const float bias = b_ih[n];
  float* o0 = xp + (size_t)(m0 + hi * 8) * Gd + n;
  float* o1 = xp + (size_t)(m0 + 16 + hi * 8) * Gd + n;
#pragma unroll
  for (int v = 0; v < 8; ++v) {
    o0[(size_t)v * Gd] = acc0[v] + bias;
    o1[(size_t)v * Gd] = acc1[v] + bias;
  }
}

// ---------------------------- grid barrier ---------------------------------

__device__ inline void grid_sync(unsigned* __restrict__ counter,
                                 unsigned* __restrict__ gen, unsigned nblocks) {
  __syncthreads();
  if (threadIdx.x == 0) {
    __threadfence();
    const unsigned g = __atomic_load_n(gen, __ATOMIC_ACQUIRE);
    if (__atomic_fetch_add(counter, 1u, __ATOMIC_ACQ_REL) == nblocks - 1) {
      __atomic_store_n(counter, 0u, __ATOMIC_RELAXED);
      __atomic_fetch_add(gen, 1u, __ATOMIC_ACQ_REL);
    } else {
      while (__atomic_load_n(gen, __ATOMIC_ACQUIRE) == g) __builtin_amdgcn_s_sleep(8);
    }
    __threadfence();
  }
  __syncthreads();
}

// ------------------------------ phase 3 ------------------------------------
// Persistent scan, 256 dependent steps. Per step:
//   async-stage h_bf (128 KB) -> LDS  | syncthreads
//   hp = h @ Whh^T : 384 N-pair tiles over 384 waves (1 pair each)
//   grid_sync | fused gates -> h, h_bf, out | grid_sync
#define NB3 48
__global__ __launch_bounds__(256) void k_rnn_scan(
    const __bf16* __restrict__ Whh, const float* __restrict__ b_hh,
    const float* __restrict__ xp, float* __restrict__ h,
    __bf16* __restrict__ h_bf, float* __restrict__ hp,
    float* __restrict__ out, unsigned* __restrict__ bar) {
  __shared__ __bf16 ldsH[Bb * Hd];                    // 128 KB of 320 KB WGP LDS

  const int tid      = threadIdx.x;
  const int gthread  = blockIdx.x * 256 + tid;
  const int wave_gid = gthread >> 5;                  // 0..383
  const int lane     = tid & 31;
  const int lo16     = lane & 15;
  const int hi       = lane >> 4;

  // wave -> (m-tile, n-pair): 4 M-tiles x 96 N-pairs = 384 pairs exactly.
  const int m0 = (wave_gid & 3) * 16;
  const int n0 = (wave_gid >> 2) * 32;
  const __bf16* arow  = ldsH + (size_t)(m0 + lo16) * Hd + hi * 16;
  const __bf16* brow0 = Whh + (size_t)(n0 + lo16) * Hd + hi * 16;
  const __bf16* brow1 = Whh + (size_t)(n0 + 16 + lo16) * Hd + hi * 16;

  for (int t = 0; t < Tt; ++t) {
    // ---- stage current h (bf16) into LDS via async loads ----------------
    stage_to_lds(ldsH, h_bf, Bb * Hd / 8, tid, 256);
    __syncthreads();

    // ---- step GEMM: hp[64x3072] = h @ Whh^T, A fragment reused ----------
    {
      v8f acc0 = {}, acc1 = {};
#pragma unroll 4
      for (int k0 = 0; k0 < Hd; k0 += 32) {
        __builtin_prefetch(brow0 + k0 + 512, 0, 1);
        __builtin_prefetch(brow1 + k0 + 512, 0, 1);
        v16bf a  = load_a_frag(arow + k0);
        v16bf b0 = *(const v16bf*)(brow0 + k0);
        v16bf b1 = *(const v16bf*)(brow1 + k0);
        acc0 = __builtin_amdgcn_wmma_f32_16x16x32_bf16(false, a, false, b0,
                                                       (short)0, acc0, false, false);
        acc1 = __builtin_amdgcn_wmma_f32_16x16x32_bf16(false, a, false, b1,
                                                       (short)0, acc1, false, false);
      }
      float* o0 = hp + (size_t)(m0 + hi * 8) * Gd + n0 + lo16;
      float* o1 = o0 + 16;
#pragma unroll
      for (int v = 0; v < 8; ++v) {
        o0[(size_t)v * Gd] = acc0[v];
        o1[(size_t)v * Gd] = acc1[v];
      }
    }

    grid_sync(bar + 0, bar + 1, NB3);

    // ---- fused gates ----------------------------------------------------
    for (int idx = gthread; idx < Bb * Hd; idx += NB3 * 256) {
      const int b = idx >> 10;
      const int i = idx & (Hd - 1);
      const float* xprow = xp + ((size_t)t * Bb + b) * Gd;
      const float* hprow = hp + (size_t)b * Gd;
      const float xr = xprow[i];
      const float xz = xprow[Hd + i];
      const float xn = xprow[2 * Hd + i];
      const float hr = hprow[i]          + b_hh[i];
      const float hz = hprow[Hd + i]     + b_hh[Hd + i];
      const float hn = hprow[2 * Hd + i] + b_hh[2 * Hd + i];
      const float r  = 1.0f / (1.0f + __expf(-(xr + hr)));
      const float z  = 1.0f / (1.0f + __expf(-(xz + hz)));
      const float nn = tanhf(xn + r * hn);
      const float hv = (1.0f - z) * nn + z * h[idx];
      h[idx]    = hv;
      h_bf[idx] = (__bf16)hv;                 // next step's staged A operand
      out[((size_t)b * Tt + t) * Hd + i] = hv;
    }

    grid_sync(bar + 0, bar + 1, NB3);
  }
}

// ------------------------------ launcher -----------------------------------

extern "C" void kernel_launch(void* const* d_in, const int* in_sizes, int n_in,
                              void* d_out, int out_size, void* d_ws, size_t ws_size,
                              hipStream_t stream) {
  (void)in_sizes; (void)n_in; (void)out_size; (void)ws_size;

  const int*   tokens = (const int*)  d_in[0];   // [T,B]
  const float* h0     = (const float*)d_in[1];   // [B,H]
  const float* emb    = (const float*)d_in[2];   // [V,H]
  const float* W_ih   = (const float*)d_in[3];   // [3H,H]
  const float* W_hh   = (const float*)d_in[4];   // [3H,H]
  const float* b_ih   = (const float*)d_in[5];   // [3H]
  const float* b_hh   = (const float*)d_in[6];   // [3H]
  float*       out    = (float*)d_out;           // [B,T,H]

  char* ws = (char*)d_ws;
  size_t off = 0;
  auto carve = [&](size_t bytes) { void* p = ws + off; off += (bytes + 255) & ~(size_t)255; return p; };
  __bf16*   x_bf   = (__bf16*)  carve((size_t)Mtot * Hd * 2);   //  32 MB
  __bf16*   Wih_bf = (__bf16*)  carve((size_t)Gd * Hd * 2);     //   6 MB
  __bf16*   Whh_bf = (__bf16*)  carve((size_t)Gd * Hd * 2);     //   6 MB
  float*    xp     = (float*)   carve((size_t)Mtot * Gd * 4);   // 192 MB
  float*    h      = (float*)   carve((size_t)Bb * Hd * 4);
  __bf16*   h_bf   = (__bf16*)  carve((size_t)Bb * Hd * 2);
  float*    hp     = (float*)   carve((size_t)Bb * Gd * 4);
  unsigned* bar    = (unsigned*)carve(64);

  k_embed_relu_bf16<<<Mtot, 256, 0, stream>>>(tokens, emb, x_bf);
  k_cvt_bf16<<<(Gd * Hd / 4 + 255) / 256, 256, 0, stream>>>(W_ih, Wih_bf, Gd * Hd / 4);
  k_cvt_bf16<<<(Gd * Hd / 4 + 255) / 256, 256, 0, stream>>>(W_hh, Whh_bf, Gd * Hd / 4);
  k_init_h<<<Bb * Hd / 256, 256, 0, stream>>>(h0, h, h_bf, bar);

  k_xp_gemm<<<dim3(Mtot / 32, Gd / 128), 256, 0, stream>>>(x_bf, Wih_bf, b_ih, xp);

  k_rnn_scan<<<NB3, 256, 0, stream>>>(Whh_bf, b_hh, xp, h, h_bf, hp, out, bar);
}
